// DMI_21680994910811
// MI455X (gfx1250) — matrix operand order repaired and checked by reading, not compiled, not run on hardware
//
#include <hip/hip_runtime.h>

typedef _Float16 h16;
typedef __attribute__((ext_vector_type(16))) _Float16 v16h;
typedef __attribute__((ext_vector_type(8)))  _Float16 h8;
typedef __attribute__((ext_vector_type(8)))  float    v8f;
typedef __attribute__((ext_vector_type(8)))  float    f8;

#define DMODEL 768
#define SEQ    2048
#define KB     128            // attention key-block size

// ---------------------------------------------------------------------------
// WMMA helper: D = A(16x32 f16) * B(32x16 f16) + C(16x16 f32)
// ---------------------------------------------------------------------------
__device__ __forceinline__ v8f wmma_f16(v16h a, v16h b, v8f c) {
    return __builtin_amdgcn_wmma_f32_16x16x32_f16(
        /*neg_a=*/false, a, /*neg_b=*/false, b,
        /*c_mod=*/(short)0, c, /*reuse_a=*/false, /*reuse_b=*/false);
}

// A-operand (16x32, MxK) per ISA layout:
//   lanes 0-15  : row M=lane,    K = {0..7, 16..23}
//   lanes 16-31 : row M=lane-16, K = {8..15, 24..31}
// p points at (row 0, k0) of a row-major [M][ld] f16 matrix.
__device__ __forceinline__ v16h ld_a16(const h16* p, int ld, int lane) {
    p += (size_t)(lane & 15) * ld + ((lane & 16) ? 8 : 0);
    h8 lo = *(const h8*)p;
    h8 hi = *(const h8*)(p + 16);
    return __builtin_shufflevector(lo, hi, 0,1,2,3,4,5,6,7,8,9,10,11,12,13,14,15);
}

// B-operand (32x16, KxN) per ISA layout (dense 16-bit B):
//   lanes 0-15  : col N=lane,    K = {0..7, 16..23}
//   lanes 16-31 : col N=lane-16, K = {8..15, 24..31}
// p points at (k0, col 0) of B stored as B^T row-major (W[n][k], k[s][d],
// vT[d][s] are all stored this way: N rows with contiguous K).
__device__ __forceinline__ v16h ld_b16(const h16* p, int ld, int lane) {
    p += (size_t)(lane & 15) * ld + ((lane & 16) ? 8 : 0);
    h8 lo = *(const h8*)p;
    h8 hi = *(const h8*)(p + 16);
    return __builtin_shufflevector(lo, hi, 0,1,2,3,4,5,6,7,8,9,10,11,12,13,14,15);
}

__device__ __forceinline__ float sigmoidf_dev(float x) {
    return 1.0f / (1.0f + __expf(-x));
}

// ---------------------------------------------------------------------------
// Elementwise fp32 -> f16 (vectorized: 8 elems/thread)
// ---------------------------------------------------------------------------
__global__ __launch_bounds__(256)
void cvt_f32_to_f16(const float* __restrict__ in, h16* __restrict__ out, int n) {
    const int i = (blockIdx.x * 256 + threadIdx.x) * 8;
    if (i < n) {
        *(h8*)(out + i) = __builtin_convertvector(*(const f8*)(in + i), h8);
    }
}

// ---------------------------------------------------------------------------
// Fused GEMM: C[m,n] = epilogue( sum_d A[m,d] * W[n,d] + bias[n] )
// M=16384, N=K=768, A and W both f16 (pre-converted). 256 threads = 8 waves;
// block tile 128x128, wave tile 32x64 (2x4 WMMA accumulators).
// EPI: 0 = f16 out (k proj)        1 = f16 out transposed vT[b][d][s]
//      2 = relu -> f16 out (gate1) 3 = sigmoid(sigmoid(x))*att*scale -> f32
//      4 = f16 out pre-scaled by 1/sqrt(D) (q proj)
// ---------------------------------------------------------------------------
template<int EPI>
__global__ __launch_bounds__(256)
void gemm_fused(const h16* __restrict__ A, const h16* __restrict__ W,
                const float* __restrict__ bias, void* __restrict__ Cp,
                const float* __restrict__ att, const float* __restrict__ tsc) {
    const int lane = threadIdx.x & 31;
    const int wid  = threadIdx.x >> 5;
    const int m0 = blockIdx.x * 128 + (wid & 3) * 32;   // 4 waves along M
    const int n0 = blockIdx.y * 128 + (wid >> 2) * 64;  // 2 waves along N

    v8f acc[2][4] = {};

    for (int k = 0; k < DMODEL; k += 32) {
        v16h a[2];
        a[0] = ld_a16(A + (size_t)m0 * DMODEL + k, DMODEL, lane);
        a[1] = ld_a16(A + (size_t)(m0 + 16) * DMODEL + k, DMODEL, lane);
        v16h b[4];
        #pragma unroll
        for (int j = 0; j < 4; ++j)
            b[j] = ld_b16(W + (size_t)(n0 + j * 16) * DMODEL + k, DMODEL, lane);
        #pragma unroll
        for (int i = 0; i < 2; ++i)
            #pragma unroll
            for (int j = 0; j < 4; ++j)
                acc[i][j] = wmma_f16(a[i], b[j], acc[i][j]);
    }

    // C/D layout: VGPR r -> row (r + 0/8 by lane-half), col = lane&15
    const float qscale = 0.036084391824351615f;  // 1/sqrt(768)
    const int col  = lane & 15;
    const int rsel = (lane & 16) ? 8 : 0;
    #pragma unroll
    for (int j = 0; j < 4; ++j) {
        const int n = n0 + j * 16 + col;
        const float bv = bias[n];
        #pragma unroll
        for (int i = 0; i < 2; ++i) {
            const int rowbase = m0 + i * 16 + rsel;
            #pragma unroll
            for (int r = 0; r < 8; ++r) {
                const int row = rowbase + r;
                const float v = acc[i][j][r] + bv;
                if (EPI == 0) {
                    ((h16*)Cp)[(size_t)row * DMODEL + n] = (h16)v;
                } else if (EPI == 1) {            // vT[b][d][s]
                    const int bb = row >> 11, ss = row & (SEQ - 1);
                    ((h16*)Cp)[(size_t)bb * DMODEL * SEQ + (size_t)n * SEQ + ss] = (h16)v;
                } else if (EPI == 2) {
                    ((h16*)Cp)[(size_t)row * DMODEL + n] = (h16)fmaxf(v, 0.0f);
                } else if (EPI == 4) {            // q projection, pre-scaled
                    ((h16*)Cp)[(size_t)row * DMODEL + n] = (h16)(v * qscale);
                } else {                           // final gating, fp32 out
                    const float g  = sigmoidf_dev(v);
                    const float gg = sigmoidf_dev(g);
                    ((float*)Cp)[(size_t)row * DMODEL + n] =
                        gg * att[(size_t)row * DMODEL + n] * tsc[n];
                }
            }
        }
    }
}

// ---------------------------------------------------------------------------
// Flash attention: one block = (batch, 32 query rows), 8 waves.
// Per 128-key iteration: each wave computes two 16x16 score subtiles
// (K-loop over d=768), scores -> LDS, online softmax, P(f16) -> LDS,
// then every wave accumulates its 96-wide dim slice of attended += P*V.
// q is pre-scaled by 1/sqrt(D) in its projection epilogue.
// Epilogue emits attended in fp32 (final gating multiply) and f16 (gate-1 A).
// ---------------------------------------------------------------------------
__global__ __launch_bounds__(256)
void attn_flash(const h16* __restrict__ qf, const h16* __restrict__ kf,
                const h16* __restrict__ vT, float* __restrict__ attO,
                h16* __restrict__ attO16) {
    __shared__ __align__(16) float sS[32 * KB];   // 16 KB
    __shared__ __align__(16) h16   sP[32 * KB];   // 8 KB
    __shared__ float sM[32], sL[32], sCorr[32];
    __shared__ float sRed[32][8];

    const int lane = threadIdx.x & 31;
    const int wid  = threadIdx.x >> 5;
    const int b    = blockIdx.y;
    const int q0   = blockIdx.x * 32;

    const h16* qbase = qf + ((size_t)b * SEQ + q0) * DMODEL;
    const h16* kbase = kf + (size_t)b * SEQ * DMODEL;
    const h16* vtb   = vT + (size_t)b * DMODEL * SEQ;

    if (threadIdx.x < 32) { sM[threadIdx.x] = -1e30f; sL[threadIdx.x] = 0.0f; }

    v8f acc[2][6] = {};                 // 32 rows x 96 dims per wave
    const int mt = wid & 1;             // score subtile row half
    const int nt = wid >> 1;            // score subtile col group (0..3)
    const int d0 = wid * 96;            // PV dim slice
    __syncthreads();

    for (int kb = 0; kb < SEQ; kb += KB) {
        // ---- scores: two 16x16 subtiles per wave (cols nt and nt+4)
        #pragma unroll
        for (int st = 0; st < 2; ++st) {
            const int ntw = nt + st * 4;
            v8f sc = {};
            for (int k = 0; k < DMODEL; k += 32) {
                v16h a  = ld_a16(qbase + (size_t)(mt * 16) * DMODEL + k, DMODEL, lane);
                v16h bm = ld_b16(kbase + (size_t)(kb + ntw * 16) * DMODEL + k,
                                 DMODEL, lane);
                sc = wmma_f16(a, bm, sc);
            }
            const int colS  = ntw * 16 + (lane & 15);
            const int rbase = mt * 16 + ((lane & 16) ? 8 : 0);
            #pragma unroll
            for (int r = 0; r < 8; ++r) sS[(rbase + r) * KB + colS] = sc[r];
        }
        __syncthreads();

        // ---- online softmax update (8 threads per row, 16 elems each)
        const int row = threadIdx.x >> 3, seg = threadIdx.x & 7;
        float vals[16]; float mx = -1e30f;
        #pragma unroll
        for (int i = 0; i < 16; ++i) {
            vals[i] = sS[row * KB + seg * 16 + i];
            mx = fmaxf(mx, vals[i]);
        }
        sRed[row][seg] = mx;
        __syncthreads();
        if (threadIdx.x < 32) {
            float tm = sRed[threadIdx.x][0];
            #pragma unroll
            for (int i = 1; i < 8; ++i) tm = fmaxf(tm, sRed[threadIdx.x][i]);
            const float mold = sM[threadIdx.x];
            const float mnew = fmaxf(mold, tm);
            const float c    = __expf(mold - mnew);
            sM[threadIdx.x] = mnew; sCorr[threadIdx.x] = c;
            sL[threadIdx.x] *= c;
        }
        __syncthreads();
        {
            const float mnew = sM[row];
            float sum = 0.0f;
            #pragma unroll
            for (int i = 0; i < 16; ++i) {
                const float p = __expf(vals[i] - mnew);
                sP[row * KB + seg * 16 + i] = (h16)p;
                sum += p;
            }
            sRed[row][seg] = sum;
        }
        __syncthreads();
        if (threadIdx.x < 32) {
            float s = 0.0f;
            #pragma unroll
            for (int i = 0; i < 8; ++i) s += sRed[threadIdx.x][i];
            sL[threadIdx.x] += s;
        }
        __syncthreads();

        // ---- rescale accumulators, then attended += P * V
        {
            const int rsel = (lane & 16) ? 8 : 0;
            #pragma unroll
            for (int i = 0; i < 2; ++i) {
                float cr[8];
                #pragma unroll
                for (int r = 0; r < 8; ++r) cr[r] = sCorr[i * 16 + rsel + r];
                #pragma unroll
                for (int j = 0; j < 6; ++j)
                    #pragma unroll
                    for (int r = 0; r < 8; ++r) acc[i][j][r] *= cr[r];
            }
        }
        #pragma unroll
        for (int kt = 0; kt < 4; ++kt) {
            v16h a0 = ld_a16((const h16*)sP + kt * 32, KB, lane);
            v16h a1 = ld_a16((const h16*)sP + 16 * KB + kt * 32, KB, lane);
            #pragma unroll
            for (int j = 0; j < 6; ++j) {
                v16h bm = ld_b16(vtb + (size_t)(d0 + j * 16) * SEQ + kb + kt * 32,
                                 SEQ, lane);
                acc[0][j] = wmma_f16(a0, bm, acc[0][j]);
                acc[1][j] = wmma_f16(a1, bm, acc[1][j]);
            }
        }
        __syncthreads();
    }

    // ---- epilogue: divide by softmax denominator, store attended (f32 + f16)
    const int colsel = lane & 15, rsel = (lane & 16) ? 8 : 0;
    #pragma unroll
    for (int i = 0; i < 2; ++i) {
        #pragma unroll
        for (int r = 0; r < 8; ++r) {
            const int lrow = i * 16 + rsel + r;
            const float inv = 1.0f / sL[lrow];
            const size_t rbase = ((size_t)(b * SEQ + q0 + lrow)) * DMODEL + d0;
            #pragma unroll
            for (int j = 0; j < 6; ++j) {
                const float v = acc[i][j][r] * inv;
                attO[rbase + j * 16 + colsel]   = v;
                attO16[rbase + j * 16 + colsel] = (h16)v;
            }
        }
    }
}

// ---------------------------------------------------------------------------
extern "C" void kernel_launch(void* const* d_in, const int* in_sizes, int n_in,
                              void* d_out, int out_size, void* d_ws, size_t ws_size,
                              hipStream_t stream) {
    const float* query = (const float*)d_in[0];
    const float* key_i = (const float*)d_in[1];
    const float* value = (const float*)d_in[2];
    const float* Wq  = (const float*)d_in[3];  const float* bq  = (const float*)d_in[4];
    const float* Wk  = (const float*)d_in[5];  const float* bk  = (const float*)d_in[6];
    const float* Wv  = (const float*)d_in[7];  const float* bv  = (const float*)d_in[8];
    const float* Wg1 = (const float*)d_in[9];  const float* bg1 = (const float*)d_in[10];
    const float* Wg2 = (const float*)d_in[11]; const float* bg2 = (const float*)d_in[12];
    const float* tsc = (const float*)d_in[13];

    const size_t M    = (size_t)8 * SEQ;            // 16384 rows
    const int    nAct = (int)(M * DMODEL);          // 12.58M elements
    const int    nW   = DMODEL * DMODEL;            // 590K elements
    const size_t szH  = M * DMODEL * sizeof(h16);   // 24 MB
    const size_t szF  = M * DMODEL * sizeof(float); // 50 MB
    const size_t szWH = (size_t)nW * sizeof(h16);   // 1.2 MB

    char* ws = (char*)d_ws;
    h16* xq16 = (h16*)(ws);                 // converted inputs (dead after proj)
    h16* xk16 = (h16*)(ws + szH);
    h16* xv16 = (h16*)(ws + 2 * szH);
    h16* qf   = (h16*)(ws + 3 * szH);
    h16* kf   = (h16*)(ws + 4 * szH);
    h16* vT   = (h16*)(ws + 5 * szH);
    float* att = (float*)(ws + 6 * szH);
    h16* w16[5];
    for (int i = 0; i < 5; ++i) w16[i] = (h16*)(ws + 6 * szH + szF + i * szWH);
    // alias dead input-conversion buffers:
    h16* att16 = xq16;                      // attended f16 (gate-1 A operand)
    h16* hbf   = xv16;                      // gate hidden f16

    dim3 blk(256);

    // --- one-shot fp32 -> f16 conversions (inputs + all weights)
    cvt_f32_to_f16<<<dim3(nAct / 8 / 256), blk, 0, stream>>>(query, xq16, nAct);
    cvt_f32_to_f16<<<dim3(nAct / 8 / 256), blk, 0, stream>>>(key_i, xk16, nAct);
    cvt_f32_to_f16<<<dim3(nAct / 8 / 256), blk, 0, stream>>>(value, xv16, nAct);
    cvt_f32_to_f16<<<dim3(nW / 8 / 256), blk, 0, stream>>>(Wq,  w16[0], nW);
    cvt_f32_to_f16<<<dim3(nW / 8 / 256), blk, 0, stream>>>(Wk,  w16[1], nW);
    cvt_f32_to_f16<<<dim3(nW / 8 / 256), blk, 0, stream>>>(Wv,  w16[2], nW);
    cvt_f32_to_f16<<<dim3(nW / 8 / 256), blk, 0, stream>>>(Wg1, w16[3], nW);
    cvt_f32_to_f16<<<dim3(nW / 8 / 256), blk, 0, stream>>>(Wg2, w16[4], nW);

    dim3 gg(128, 6);     // 16384/128 x 768/128

    // Projections (q pre-scaled by 1/sqrt(D); v written transposed for PV)
    gemm_fused<4><<<gg, blk, 0, stream>>>(xq16, w16[0], bq, qf, nullptr, nullptr);
    gemm_fused<0><<<gg, blk, 0, stream>>>(xk16, w16[1], bk, kf, nullptr, nullptr);
    gemm_fused<1><<<gg, blk, 0, stream>>>(xv16, w16[2], bv, vT, nullptr, nullptr);

    // Flash attention -> attended (fp32 + f16)
    dim3 ga(SEQ / 32, 8);
    attn_flash<<<ga, blk, 0, stream>>>(qf, kf, vT, att, att16);

    // Gate: h = relu(att @ Wg1^T + bg1)  (f16)
    gemm_fused<2><<<gg, blk, 0, stream>>>(att16, w16[3], bg1, hbf, nullptr, nullptr);
    // out = sigmoid(sigmoid(h @ Wg2^T + bg2)) * att * text_scale  (fp32)
    gemm_fused<3><<<gg, blk, 0, stream>>>(hbf, w16[4], bg2, d_out, att, tsc);
}
